// HierachicalEncoder_58196806861037
// MI455X (gfx1250) — matrix operand order repaired and checked by reading, not compiled, not run on hardware
//
#include <hip/hip_runtime.h>
#include <math.h>
#include <stdint.h>

typedef _Float16 h16;
typedef __attribute__((ext_vector_type(16))) _Float16 v16h;
typedef __attribute__((ext_vector_type(8)))  _Float16 v8h;
typedef __attribute__((ext_vector_type(8)))  float    v8f;
typedef __attribute__((address_space(3)))    h16      h16_lds;

#define B_DIM 16
#define N_DIM 2048
#define E_DIM 1024

// Use GLOBAL_LOAD_ASYNC_TO_LDS_B128 + s_wait_asynccnt for the y-strip staging.
#define USE_ASYNC_LDS 1

#define WMMA_F16(a, b, c) \
    __builtin_amdgcn_wmma_f32_16x16x32_f16(false, (a), false, (b), (short)0, (c), false, false)

// ---- WMMA fragment loaders (contiguous k order k = 16*(lane/16) + s, same for A and B) ----

// A 16x32 f16: src row-major, row stride ld halves, 16 rows starting at src.
// lanes 0-15: M=lane, halves 0..15 -> K=kb..kb+15
// lanes 16-31: same M, K=kb+16..kb+31.
__device__ inline v16h load_a_frag(const h16* src, int ld, int kb, int lane) {
    int m  = lane & 15;
    int hi = (lane >> 4) & 1;
    const h16* p = src + (size_t)m * ld + kb + hi * 16;
    v8h lo = *(const v8h*)(p);
    v8h hc = *(const v8h*)(p + 8);
    v16h a;
#pragma unroll
    for (int i = 0; i < 8; ++i) { a[i] = lo[i]; a[i + 8] = hc[i]; }
    return a;
}

// B 32x16 f16 where B[k,j] = src[j*ld + k] (src rows are B columns).
// lanes 0-15 hold K=kb..kb+15 for N=lane; lanes 16-31 hold K=kb+16..kb+31.
// -> 16 CONSECUTIVE halves per lane (32B, two b128 loads).
__device__ inline v16h load_b_frag_rows(const h16* src, int ld, int kb, int lane) {
    int j    = lane & 15;
    int koff = kb + ((lane >> 4) & 1) * 16;
    return *(const v16h*)(src + (size_t)j * ld + koff);
}

// ---- Kernel 1: L2-normalize rows of features -> x (f16); zero colw ----
__global__ __launch_bounds__(256) void k_normalize(const float* __restrict__ f,
                                                   h16* __restrict__ x,
                                                   float* __restrict__ colw) {
    const int row = blockIdx.x;                 // 0 .. B*N-1
    const float* src = f + (size_t)row * E_DIM;
    __shared__ float red[256];
    float ss = 0.f;
    for (int e = threadIdx.x; e < E_DIM; e += 256) { float v = src[e]; ss += v * v; }
    red[threadIdx.x] = ss;
    __syncthreads();
    for (int off = 128; off > 0; off >>= 1) {
        if (threadIdx.x < off) red[threadIdx.x] += red[threadIdx.x + off];
        __syncthreads();
    }
    const float inv = 1.0f / fmaxf(sqrtf(red[0]), 1e-12f);
    h16* dst = x + (size_t)row * E_DIM;
    for (int e = threadIdx.x; e < E_DIM; e += 256) dst[e] = (h16)(src[e] * inv);
    if (threadIdx.x == 0) colw[row] = 0.0f;
}

// ---- Kernel 2: Mt = (W_q^T @ W_k)^T stored TRANSPOSED (Mt[j,i] = M[i,j]).
//      f32 in, f16 out. One 16x16 tile per wave; tiny op (2 GFLOP). ----
__global__ __launch_bounds__(256) void k_mmat(const float* __restrict__ wq,
                                              const float* __restrict__ wk,
                                              h16* __restrict__ Mt) {
    const int wave = threadIdx.x >> 5, lane = threadIdx.x & 31;
    const int tile = blockIdx.x * 8 + wave;     // 0 .. 4095
    const int i0 = (tile >> 6) * 16;
    const int j0 = (tile & 63) * 16;
    const int i  = lane & 15;
    const int hi = (lane >> 4) & 1;
    v8f c = {};
    for (int kb = 0; kb < E_DIM; kb += 32) {
        v16h a, b;
#pragma unroll
        for (int t = 0; t < 8; ++t) {          // A[i,k] = wq[k*E + i0+i]
            a[t]     = (h16)wq[(size_t)(kb + hi * 16 + t) * E_DIM + i0 + i];
            a[t + 8] = (h16)wq[(size_t)(kb + hi * 16 + 8 + t) * E_DIM + i0 + i];
        }
        const int kbase = kb + hi * 16;
#pragma unroll
        for (int t = 0; t < 16; ++t)           // B[k,j] = wk[k*E + j0+j]
            b[t] = (h16)wk[(size_t)(kbase + t) * E_DIM + j0 + i];
        c = WMMA_F16(a, b, c);
    }
#pragma unroll
    for (int r = 0; r < 8; ++r)                // M[i0+r+8hi, j0+i] -> Mt[col*E + row]
        Mt[(size_t)(j0 + i) * E_DIM + i0 + r + 8 * hi] = (h16)c[r];
}

// ---- Kernel 3: y = x @ M  ([32768,1024] @ [1024,1024]) with B = Mt rows.
//      No LDS, no syncs: wave tile 16x64, 4 WMMAs reuse one A-frag per K-step. ----
__global__ __launch_bounds__(256) void k_gemm_y(const h16* __restrict__ x,
                                                const h16* __restrict__ Mt,
                                                h16* __restrict__ y) {
    const int wave = threadIdx.x >> 5, lane = threadIdx.x & 31;
    const int rt   = (blockIdx.x >> 4) * 8 + wave;  // 0..2047 row tiles
    const int col0 = (blockIdx.x & 15) * 64;        // 16 col groups
    const int row0 = rt * 16;
    const h16* arow = x + (size_t)row0 * E_DIM;
    v8f c0 = {}, c1 = {}, c2 = {}, c3 = {};
    for (int kb = 0; kb < E_DIM; kb += 32) {
        v16h a  = load_a_frag(arow, E_DIM, kb, lane);
        v16h b0 = load_b_frag_rows(Mt + (size_t)(col0     ) * E_DIM, E_DIM, kb, lane);
        v16h b1 = load_b_frag_rows(Mt + (size_t)(col0 + 16) * E_DIM, E_DIM, kb, lane);
        v16h b2 = load_b_frag_rows(Mt + (size_t)(col0 + 32) * E_DIM, E_DIM, kb, lane);
        v16h b3 = load_b_frag_rows(Mt + (size_t)(col0 + 48) * E_DIM, E_DIM, kb, lane);
        c0 = WMMA_F16(a, b0, c0);
        c1 = WMMA_F16(a, b1, c1);
        c2 = WMMA_F16(a, b2, c2);
        c3 = WMMA_F16(a, b3, c3);
    }
    const int hi = (lane >> 4) & 1, nn = lane & 15;
#pragma unroll
    for (int rr = 0; rr < 8; ++rr) {
        const size_t orow = (size_t)(row0 + rr + 8 * hi) * E_DIM + col0 + nn;
        y[orow]      = (h16)c0[rr];
        y[orow + 16] = (h16)c1[rr];
        y[orow + 32] = (h16)c2[rr];
        y[orow + 48] = (h16)c3[rr];
    }
}

// ---- Kernel 4: per (b, 16-row strip): S = y_strip @ x[b]^T * E^-.5, row softmax,
//      accumulate column means into colw. Score strip (16x2048 f32) lives in LDS. ----
__global__ __launch_bounds__(256) void k_attn(const h16* __restrict__ x,
                                              const h16* __restrict__ y,
                                              float* __restrict__ colw) {
    __shared__ h16  yA[16 * E_DIM];             // 32 KB: 16 rows of y
    __shared__ float sS[16 * N_DIM];            // 128 KB: full 16x2048 score strip
    __shared__ float red[256];
    __shared__ float rowmax[16], rowinv[16];
    const int tid = threadIdx.x, wave = tid >> 5, lane = tid & 31;
    const int b  = blockIdx.x >> 7;             // N/16 = 128 strips per batch
    const int n0 = (blockIdx.x & 127) << 4;
    const h16* xb   = x + (size_t)b * N_DIM * E_DIM;
    const h16* ysrc = y + (size_t)b * N_DIM * E_DIM + (size_t)n0 * E_DIM;

    // stage 16 contiguous y rows (32 KB) into LDS
#if USE_ASYNC_LDS
    {
        // LDS byte offset of yA via addrspace(3) ptrtoint; async copy 16B/lane,
        // tracked by ASYNCcnt (CDNA5 ISA).
        const uint32_t lds0 = (uint32_t)(uintptr_t)(h16_lds*)yA;
        for (int i = tid * 8; i < 16 * E_DIM; i += 256 * 8) {
            uint32_t laddr = lds0 + (uint32_t)(i * 2);
            uint64_t gaddr = (uint64_t)(uintptr_t)(ysrc + i);
            asm volatile("global_load_async_to_lds_b128 %0, %1, off"
                         :: "v"(laddr), "v"(gaddr) : "memory");
        }
        asm volatile("s_wait_asynccnt 0x0" ::: "memory");
    }
#else
    for (int i = tid * 8; i < 16 * E_DIM; i += 256 * 8)
        *(v8h*)&yA[i] = *(const v8h*)&ysrc[i];
#endif
    __syncthreads();

    const float scale = 0.03125f;               // 1024^-0.5
    const int hi = (lane >> 4) & 1, nn = lane & 15;
    // 128 m-tiles; each wave does 4 passes of 4 tiles (16x64 wave tile, A reused 4x)
    for (int p = 0; p < 4; ++p) {
        const int m0 = (wave * 16 + p * 4) * 16;
        v8f c0 = {}, c1 = {}, c2 = {}, c3 = {};
        for (int kb = 0; kb < E_DIM; kb += 32) {
            v16h a  = load_a_frag(yA, E_DIM, kb, lane);
            v16h b0 = load_b_frag_rows(xb + (size_t)(m0     ) * E_DIM, E_DIM, kb, lane);
            v16h b1 = load_b_frag_rows(xb + (size_t)(m0 + 16) * E_DIM, E_DIM, kb, lane);
            v16h b2 = load_b_frag_rows(xb + (size_t)(m0 + 32) * E_DIM, E_DIM, kb, lane);
            v16h b3 = load_b_frag_rows(xb + (size_t)(m0 + 48) * E_DIM, E_DIM, kb, lane);
            c0 = WMMA_F16(a, b0, c0);
            c1 = WMMA_F16(a, b1, c1);
            c2 = WMMA_F16(a, b2, c2);
            c3 = WMMA_F16(a, b3, c3);
        }
#pragma unroll
        for (int rr = 0; rr < 8; ++rr) {
            float* srow = &sS[(rr + 8 * hi) * N_DIM + m0 + nn];
            srow[0]  = c0[rr] * scale;
            srow[16] = c1[rr] * scale;
            srow[32] = c2[rr] * scale;
            srow[48] = c3[rr] * scale;
        }
    }
    __syncthreads();

    // softmax per row: 16 threads per row
    const int row = tid >> 4, sub = tid & 15;
    float mx = -3.0e38f;
    for (int m = sub; m < N_DIM; m += 16) mx = fmaxf(mx, sS[row * N_DIM + m]);
    red[row * 16 + sub] = mx;
    __syncthreads();
    if (sub == 0) {
        float v = red[row * 16];
        for (int t = 1; t < 16; ++t) v = fmaxf(v, red[row * 16 + t]);
        rowmax[row] = v;
    }
    __syncthreads();
    const float rm = rowmax[row];
    float sum = 0.f;
    for (int m = sub; m < N_DIM; m += 16) {
        float e = __expf(sS[row * N_DIM + m] - rm);
        sS[row * N_DIM + m] = e;
        sum += e;
    }
    red[row * 16 + sub] = sum;
    __syncthreads();
    if (sub == 0) {
        float v = 0.f;
        for (int t = 0; t < 16; ++t) v += red[row * 16 + t];
        rowinv[row] = 1.0f / v;
    }
    __syncthreads();
    // column reduction: colw[b,m] += (1/N) * sum_r exp_r(m) / sumexp_r
    for (int m = tid; m < N_DIM; m += 256) {
        float acc = 0.f;
#pragma unroll
        for (int rr = 0; rr < 16; ++rr) acc += sS[rr * N_DIM + m] * rowinv[rr];
        atomicAdd(&colw[b * N_DIM + m], acc * (1.0f / N_DIM));
    }
}

// ---- Kernel 5: out[b,e] = sum_m colw[b,m] * x[b,m,e] ----
__global__ __launch_bounds__(256) void k_out(const h16* __restrict__ x,
                                             const float* __restrict__ colw,
                                             float* __restrict__ out) {
    const int b = blockIdx.x >> 2;              // E/256 = 4 chunks per batch
    const int e = ((blockIdx.x & 3) << 8) + threadIdx.x;
    const h16* xb = x + (size_t)b * N_DIM * E_DIM;
    const float* w = colw + b * N_DIM;
    float acc = 0.f;
    for (int m = 0; m < N_DIM; ++m) {
        __builtin_prefetch(&xb[(size_t)(m + 8) * E_DIM + e], 0, 3);
        acc += w[m] * (float)xb[(size_t)m * E_DIM + e];
    }
    out[b * E_DIM + e] = acc;
}

extern "C" void kernel_launch(void* const* d_in, const int* in_sizes, int n_in,
                              void* d_out, int out_size, void* d_ws, size_t ws_size,
                              hipStream_t stream) {
    const float* features = (const float*)d_in[0];   // [16,2048,1024] f32
    const float* w_q      = (const float*)d_in[1];   // [1024,1024] f32
    const float* w_k      = (const float*)d_in[2];   // [1024,1024] f32
    float* out = (float*)d_out;                      // [16,1024] f32

    const size_t XN = (size_t)B_DIM * N_DIM * E_DIM; // 33,554,432
    h16*   x_h  = (h16*)d_ws;                        // 64 MB
    h16*   y_h  = x_h + XN;                          // 64 MB
    h16*   Mt_h = y_h + XN;                          // 2 MB (M transposed)
    float* colw = (float*)(Mt_h + (size_t)E_DIM * E_DIM); // 128 KB

    k_normalize<<<B_DIM * N_DIM, 256, 0, stream>>>(features, x_h, colw);
    k_mmat<<<(E_DIM / 16) * (E_DIM / 16) / 8, 256, 0, stream>>>(w_q, w_k, Mt_h);
    k_gemm_y<<<(B_DIM * N_DIM / 16 / 8) * (E_DIM / 64), 256, 0, stream>>>(x_h, Mt_h, y_h);
    k_attn<<<B_DIM * (N_DIM / 16), 256, 0, stream>>>(x_h, y_h, colw);
    k_out<<<B_DIM * (E_DIM / 256), 256, 0, stream>>>(x_h, colw, out);
}
